// CausalMultiEdgeFunctionalGraph_52544629899448
// MI455X (gfx1250) — matrix operand order repaired and checked by reading, not compiled, not run on hardware
//
#include <hip/hip_runtime.h>
#include <hip/hip_bf16.h>

#define DECAY_F 0.99f
#define OMD_F   0.01f
#define EPS_F   1e-8f

constexpr int B_ = 8, T_ = 256, S_ = 64, F_ = 32, OUT_ = 128;

typedef __attribute__((ext_vector_type(16))) _Float16 v16h;
typedef __attribute__((ext_vector_type(8)))  _Float16 v8h;
typedef __attribute__((ext_vector_type(8)))  float    v8f;

// ---------------------------------------------------------------------------
// Kernel 1: sequential EMA-covariance scan. One block per (b,f) chain
// (256 blocks, 256 threads). Thread owns a 16-wide slice of one cov row:
// r = tid/4, cols [16*(tid%4), +16). cov + W_sym live in registers.
// The 4 threads of a row are consecutive lanes -> row reductions are
// wave shuffles, leaving only 2 LDS barriers per step:
//   (A) publish x_c / std,  (B) publish 1/sqrt(D_hat).
// Output x_g is written as f16 (it feeds the f16 WMMA A-operand anyway).
// ---------------------------------------------------------------------------
__global__ __launch_bounds__(256) void scan_kernel(
    const float* __restrict__ x_sources,   // (B,T,S,F) f32
    const float* __restrict__ W_graph,     // (F,S,S)   f32
    _Float16* __restrict__ x_g)            // (B,T,S,F) f16 workspace
{
    const int bf  = blockIdx.x;
    const int b   = bf >> 5;        // / F_
    const int f   = bf & (F_ - 1);
    const int tid = threadIdx.x;
    const int r   = tid >> 2;       // row 0..63
    const int q   = tid & 3;        // column quarter (consecutive lanes!)
    const int c0  = q * 16;

    __shared__ float xc_s[S_];
    __shared__ float std_s[S_];
    __shared__ float invsqD_s[S_];

    // Symmetrized graph weights for our 16 cov elements (constant over t).
    float wsym[16];
    const float* Wg = W_graph + (size_t)f * S_ * S_;
#pragma unroll
    for (int j = 0; j < 16; ++j) {
        const int c = c0 + j;
        wsym[j] = 0.5f * (Wg[r * S_ + c] + Wg[c * S_ + r]);
    }

    float cov[16];
#pragma unroll
    for (int j = 0; j < 16; ++j) cov[j] = 0.0f;

    const float* xb = x_sources + (size_t)b * T_ * S_ * F_ + f;
    _Float16*    og = x_g       + (size_t)b * T_ * S_ * F_ + f;

    // Per-row state held by threads 0..63 (s = tid): mean + diagonal cov.
    float mu   = 0.0f;
    float covd = 0.0f;
    float xv   = (tid < S_) ? xb[(size_t)tid * F_] : 0.0f;   // x at t=0

    float decay_pow = 1.0f;
    for (int t = 0; t < T_; ++t) {
        decay_pow *= DECAY_F;
        const float bias_corr = 1.0f - decay_pow;
        const float inv_bc    = 1.0f / bias_corr;

        // ---- step 1: mean update, centered x, row std (threads 0..63) ----
        if (tid < S_) {
            mu = DECAY_F * mu + OMD_F * xv;
            const float xc = xv - mu * inv_bc;
            covd = DECAY_F * covd + OMD_F * xc * xc;     // cov[s][s] recurrence
            xc_s[tid]  = xc;
            std_s[tid] = sqrtf(covd * inv_bc + EPS_F);
        }
        __syncthreads();                                  // barrier A

        // software-pipelined fetch of x_{t+1} (latency hidden by step 2/3)
        if (tid < S_ && (t + 1) < T_)
            xv = xb[(size_t)((t + 1) * S_ + tid) * F_];

        // ---- step 2: cov EMA update, A_eff, row-sum D_hat ----
        const float xr = xc_s[r];
        float xcol[16];
#pragma unroll
        for (int j = 0; j < 16; ++j) xcol[j] = xc_s[c0 + j];

#pragma unroll
        for (int j = 0; j < 16; ++j)
            cov[j] = DECAY_F * cov[j] + OMD_F * xr * xcol[j];

        const float sr = std_s[r];
        float aeff[16];
        float dsum = 0.0f;
#pragma unroll
        for (int j = 0; j < 16; ++j) {
            const float cov_c = cov[j] * inv_bc;
            const float corr  = cov_c / (sr * std_s[c0 + j] + EPS_F);
            float a = corr * wsym[j];
            a = fmaxf(a, 0.0f);
            aeff[j] = a;
            dsum += a;
        }
        // reduce over the 4 column-slice lanes of this row (same wave)
        dsum += __shfl_xor(dsum, 1);
        dsum += __shfl_xor(dsum, 2);
        const float invD = rsqrtf(dsum + EPS_F);          // 1/sqrt(D_hat[r])
        if (q == 0) invsqD_s[r] = invD;
        __syncthreads();                                  // barrier B

        // ---- step 3: degree-normalized matvec, emit x_g row (f16) ----
        float mp = 0.0f;
#pragma unroll
        for (int j = 0; j < 16; ++j)
            mp += aeff[j] * xcol[j] * invsqD_s[c0 + j];
        mp += __shfl_xor(mp, 1);
        mp += __shfl_xor(mp, 2);
        if (q == 0)
            og[(size_t)(t * S_ + r) * F_] = (_Float16)(invD * mp);
        // no end-of-loop barrier needed: next writes to xc_s/std_s happen
        // after barrier B (all waves), next invsqD_s write after next A.
    }
}

// ---------------------------------------------------------------------------
// Kernel 2: out(131072,128) = x_g(131072,32) @ W_dense(32,128).
// One V_WMMA_F32_16X16X32_F16 per 16x16 tile (whole K in one op).
// Block = 8 waves; wave w -> M-tile, loops over the 8 N-tiles with A reused.
// A comes straight from the f16 workspace (two 16B loads per lane, no cvt).
// D tiles are staged in LDS (row stride 132 -> conflict-free) so the 64 MB
// output is written with contiguous global_store_b128 (16 per lane).
// ---------------------------------------------------------------------------
__global__ __launch_bounds__(256) void proj_kernel(
    const _Float16* __restrict__ x_g,   // (M=131072, 32) f16
    const float* __restrict__ W_dense,  // (32, 128) f32
    float* __restrict__ out)            // (M, 128) f32
{
    __shared__ _Float16 Wt[OUT_ * F_];          // [n][k] transposed, 8 KB
    __shared__ float tile[8][16 * 132];         // per-wave D staging, ~66 KB
    const int tid = threadIdx.x;
    for (int i = tid; i < OUT_ * F_; i += 256) {
        const int n = i >> 5;    // / F_
        const int k = i & 31;
        Wt[i] = (_Float16)W_dense[k * OUT_ + n];
    }
    __syncthreads();

    const int wave  = tid >> 5;
    const int lane  = tid & 31;
    const int lhalf = lane >> 4;
    const int lmod  = lane & 15;

    const int mtile = blockIdx.x * 8 + wave;
    const int rowm  = mtile * 16 + lmod;                 // A-matrix row
    const _Float16* arow = x_g + (size_t)rowm * F_;
    const int koff = lhalf * 8;                          // high lanes: K+8

    // A (16x32 f16): e<8 -> K=e+koff ; e>=8 -> K=e+8+koff (two 16B loads)
    const v8h alo = *(const v8h*)(arow + koff);
    const v8h ahi = *(const v8h*)(arow + 16 + koff);
    v16h a;
#pragma unroll
    for (int e = 0; e < 8; ++e) { a[e] = alo[e]; a[8 + e] = ahi[e]; }

    float* wtile = tile[wave];
#pragma unroll
    for (int nt = 0; nt < 8; ++nt) {
        const int n = nt * 16 + lmod;                    // B/D column
        const _Float16* wrow = &Wt[n * F_ + lhalf * 8];  // e<8 -> K=e+8*lhalf ; e>=8 -> K=e+8+8*lhalf
        const v8h blo = *(const v8h*)(wrow);
        const v8h bhi = *(const v8h*)(wrow + 16);
        v16h bm;
#pragma unroll
        for (int e = 0; e < 8; ++e) { bm[e] = blo[e]; bm[8 + e] = bhi[e]; }

        v8f c = {};
        c = __builtin_amdgcn_wmma_f32_16x16x32_f16(
                false, a, false, bm, (short)0, c, false, false);

        // D: VGPR v -> row m = v + 8*lhalf, col n (LDS row stride 132)
#pragma unroll
        for (int v = 0; v < 8; ++v)
            wtile[(v + 8 * lhalf) * 132 + n] = c[v];
    }

    // Coalesced writeback: this wave's tile is a contiguous 8 KB of `out`.
    float* obase = out + (size_t)mtile * 16 * OUT_;
#pragma unroll
    for (int k = 0; k < 16; ++k) {
        const float4 dv = *(const float4*)&wtile[k * 132 + lane * 4];
        *(float4*)&obase[k * OUT_ + lane * 4] = dv;
    }
}

// ---------------------------------------------------------------------------
extern "C" void kernel_launch(void* const* d_in, const int* in_sizes, int n_in,
                              void* d_out, int out_size, void* d_ws, size_t ws_size,
                              hipStream_t stream) {
    const float* x_sources = (const float*)d_in[0];  // (8,256,64,32)
    const float* W_graph   = (const float*)d_in[1];  // (32,64,64)
    const float* W_dense   = (const float*)d_in[2];  // (32,128)
    float* out = (float*)d_out;                      // (8,256,64,128)
    _Float16* x_g = (_Float16*)d_ws;                 // (8,256,64,32) f16 = 8 MB

    // 256 independent (b,f) scan chains, one block each.
    scan_kernel<<<dim3(B_ * F_), dim3(256), 0, stream>>>(x_sources, W_graph, x_g);

    // M = B*T*S = 131072 rows -> 8192 M-tiles of 16 -> 8 per block.
    const int m_tiles = (B_ * T_ * S_) / 16;
    proj_kernel<<<dim3(m_tiles / 8), dim3(256), 0, stream>>>(x_g, W_dense, out);
}
